// CrossEntropyLoss_24326694765140
// MI455X (gfx1250) — matrix operand order repaired and checked by reading, not compiled, not run on hardware
//
#include <hip/hip_runtime.h>

// ---------------------------------------------------------------------------
// Fused linear + cross-entropy for MI455X (gfx1250, wave32, WMMA).
// T=4096 tokens, D=2048, V=32000.
//
//   1) convert W, H  f32 -> bf16 (one HBM pass; bf16 W+H fit in 192MB L2)
//   2) gold[t] = dot_f32(H[t], W[label[t]])
//   3) WMMA GEMM, grid (125 vocab-slabs x 32 token-tiles), block tile
//      128 tok x 256 vocab, 8 waves of 64x64. Double-buffered LDS staged by
//      global_load_async_to_lds_b128 (ASYNCcnt), overlapped with compute.
//      Per-tile (max, sumexp) partials -> ws (logits never hit HBM).
//   4) per-token online-LSE merge of 125 slab partials, mask, block-sum
//   5) final sum * loss_weight -> d_out[0]
// ---------------------------------------------------------------------------

#define T_TOK 4096
#define DIM   2048
#define VOCAB 32000
#define NSLAB 125      // VOCAB / 256
#define LDSS  88       // LDS row stride in bf16 (176B: 16B-aligned, low-conflict)

typedef __bf16 bf16;
typedef __attribute__((ext_vector_type(16))) __bf16 v16bf;
typedef __attribute__((ext_vector_type(8)))  float  v8f;

struct Frag { union { v16bf v; uint4 q[2]; }; };

// ---------------------------------------------------------------- conversion
__global__ void cvt_f32_to_bf16(const float* __restrict__ src,
                                bf16* __restrict__ dst, int n4) {
  int i = blockIdx.x * blockDim.x + threadIdx.x;
  if (i >= n4) return;
  float4 v = ((const float4*)src)[i];
  union { bf16 h[4]; uint2 u; } o;
  o.h[0] = (bf16)v.x; o.h[1] = (bf16)v.y; o.h[2] = (bf16)v.z; o.h[3] = (bf16)v.w;
  ((uint2*)dst)[i] = o.u;
}

// ---------------------------------------------------------------- gold logit
__global__ void gold_kernel(const float* __restrict__ H,
                            const float* __restrict__ W,
                            const int* __restrict__ labels,
                            float* __restrict__ gold) {
  int wave = threadIdx.x >> 5, lane = threadIdx.x & 31;
  int t = blockIdx.x * 8 + wave;
  int lab = labels[t];
  float acc = 0.0f;
  if (lab >= 0) {
    const float* h = H + (size_t)t * DIM;
    const float* w = W + (size_t)lab * DIM;
    for (int k = lane; k < DIM; k += 32) acc = fmaf(h[k], w[k], acc);
  }
  #pragma unroll
  for (int s = 16; s > 0; s >>= 1) acc += __shfl_xor(acc, s, 32);
  if (lane == 0) gold[t] = acc;
}

// ------------------------------------------------------- WMMA GEMM + partial
// grid = (NSLAB, T/128); block 256 = 8 waves (2M x 4N), 64x64 tile per wave.
__global__ __launch_bounds__(256)
void gemm_lse_partial(const bf16* __restrict__ Hb, const bf16* __restrict__ Wb,
                      float* __restrict__ pm, float* __restrict__ ps) {
  __shared__ __align__(16) bf16 Asl[2][128 * LDSS];   // 128 tok  x 64 K
  __shared__ __align__(16) bf16 Bsl[2][256 * LDSS];   // 256 voc  x 64 K
  __shared__ float redm[128 * 4];
  __shared__ float reds[128 * 4];

  const int tid   = threadIdx.x;
  const int lane  = tid & 31;
  const int lr    = lane & 15;
  const int lh    = lane >> 4;
  const int wave  = tid >> 5;
  const int waveM = wave >> 2;     // 0..1
  const int waveN = wave & 3;      // 0..3
  const int mo    = waveM * 64;
  const int no    = waveN * 64;

  const int v0 = blockIdx.x * 256;
  const int t0 = blockIdx.y * 128;

  // LDS byte addresses (generic LDS addr truncates to LDS offset per ISA 10.2)
  const uint32_t ldsA = (uint32_t)(uintptr_t)&Asl[0][0];
  const uint32_t ldsB = (uint32_t)(uintptr_t)&Bsl[0][0];
  const unsigned long long hbase = (unsigned long long)Hb;
  const unsigned long long wbase = (unsigned long long)Wb;

  // -------- async staging of one 64-wide K chunk into buffer `buf` --------
  auto stage = [&](int buf, int kc) {
    #pragma unroll
    for (int j = 0; j < 4; ++j) {                      // A: 128x64 = 1024 b128
      int c = tid + j * 256, row = c >> 3, col = (c & 7) << 3;
      uint32_t l = ldsA + (uint32_t)buf * (128 * LDSS * 2) +
                   (uint32_t)(row * LDSS + col) * 2;
      uint32_t g = (uint32_t)(((t0 + row) * DIM + kc + col) * 2);
      asm volatile("global_load_async_to_lds_b128 %0, %1, %2"
                   :: "v"(l), "v"(g), "s"(hbase) : "memory");
    }
    #pragma unroll
    for (int j = 0; j < 8; ++j) {                      // B: 256x64 = 2048 b128
      int c = tid + j * 256, row = c >> 3, col = (c & 7) << 3;
      uint32_t l = ldsB + (uint32_t)buf * (256 * LDSS * 2) +
                   (uint32_t)(row * LDSS + col) * 2;
      uint32_t g = (uint32_t)(((v0 + row) * DIM + kc + col) * 2);
      asm volatile("global_load_async_to_lds_b128 %0, %1, %2"
                   :: "v"(l), "v"(g), "s"(wbase) : "memory");
    }
  };

  v8f acc[4][4];
  #pragma unroll
  for (int mi = 0; mi < 4; ++mi)
    #pragma unroll
    for (int ni = 0; ni < 4; ++ni)
      acc[mi][ni] = (v8f){0.f, 0.f, 0.f, 0.f, 0.f, 0.f, 0.f, 0.f};

  // prologue: fill buffer 0
  stage(0, 0);
  asm volatile("s_wait_asynccnt 0" ::: "memory");
  __syncthreads();

  int buf = 0;
  for (int kc = 0; kc < DIM; kc += 64) {
    if (kc + 64 < DIM) stage(buf ^ 1, kc + 64);        // overlap next-chunk DMA

    #pragma unroll
    for (int ks = 0; ks < 64; ks += 32) {
      Frag a[4], b[4];
      // A 16x32 bf16 (ISA 7.12.2): lanes0-15 K0-7|16-23, lanes16-31 K8-15|24-31
      #pragma unroll
      for (int mi = 0; mi < 4; ++mi) {
        const bf16* p = &Asl[buf][(mo + mi * 16 + lr) * LDSS + ks + (lh ? 8 : 0)];
        a[mi].q[0] = *(const uint4*)p;
        a[mi].q[1] = *(const uint4*)(p + 16);
      }
      // B 32x16 bf16: column n = W row; lanes0-15 K0-7|16-23, lanes16-31 K8-15|24-31
      #pragma unroll
      for (int ni = 0; ni < 4; ++ni) {
        const bf16* p = &Bsl[buf][(no + ni * 16 + lr) * LDSS + ks + (lh ? 8 : 0)];
        b[ni].q[0] = *(const uint4*)p;
        b[ni].q[1] = *(const uint4*)(p + 16);
      }
      #pragma unroll
      for (int ni = 0; ni < 4; ++ni)
        #pragma unroll
        for (int mi = 0; mi < 4; ++mi)
          acc[mi][ni] = __builtin_amdgcn_wmma_f32_16x16x32_bf16(
              false, a[mi].v, false, b[ni].v, (short)0, acc[mi][ni],
              false, false);
    }

    asm volatile("s_wait_asynccnt 0" ::: "memory");    // next buffer landed
    __syncthreads();                                   // all waves done reading
    buf ^= 1;
  }

  // ---- per-token (row) max / sumexp over this 256-vocab slab ----
  // C layout: VGPR r -> row mo+mi*16+r (lanes0-15) / +8+r (lanes16-31); lane -> col.
  #pragma unroll
  for (int mi = 0; mi < 4; ++mi) {
    #pragma unroll
    for (int r = 0; r < 8; ++r) {
      float m = fmaxf(fmaxf(acc[mi][0][r], acc[mi][1][r]),
                      fmaxf(acc[mi][2][r], acc[mi][3][r]));
      #pragma unroll
      for (int s = 1; s < 16; s <<= 1) m = fmaxf(m, __shfl_xor(m, s, 16));
      float e = __expf(acc[mi][0][r] - m) + __expf(acc[mi][1][r] - m) +
                __expf(acc[mi][2][r] - m) + __expf(acc[mi][3][r] - m);
      #pragma unroll
      for (int s = 1; s < 16; s <<= 1) e += __shfl_xor(e, s, 16);
      if (lr == r) {
        int row = mo + mi * 16 + r + (lh ? 8 : 0);
        redm[row * 4 + waveN] = m;
        reds[row * 4 + waveN] = e;
      }
    }
  }
  __syncthreads();

  if (tid < 128) {
    float m0 = redm[tid * 4 + 0], m1 = redm[tid * 4 + 1];
    float m2 = redm[tid * 4 + 2], m3 = redm[tid * 4 + 3];
    float M = fmaxf(fmaxf(m0, m1), fmaxf(m2, m3));
    float S = reds[tid * 4 + 0] * __expf(m0 - M) +
              reds[tid * 4 + 1] * __expf(m1 - M) +
              reds[tid * 4 + 2] * __expf(m2 - M) +
              reds[tid * 4 + 3] * __expf(m3 - M);
    pm[(size_t)blockIdx.x * T_TOK + (t0 + tid)] = M;
    ps[(size_t)blockIdx.x * T_TOK + (t0 + tid)] = S;
  }
}

// ---------------------------------------------------- per-token LSE + reduce
__global__ void reduce_lse(const float* __restrict__ pm,
                           const float* __restrict__ ps,
                           const float* __restrict__ gold,
                           const int* __restrict__ labels,
                           float* __restrict__ bsum) {
  int t = blockIdx.x * 256 + threadIdx.x;
  float M = -__builtin_inff(), S = 0.0f;
  for (int s = 0; s < NSLAB; ++s) {          // coalesced: [slab][token]
    float m  = pm[(size_t)s * T_TOK + t];
    float ss = ps[(size_t)s * T_TOK + t];
    if (m > M) { S = S * __expf(M - m) + ss; M = m; }
    else       { S += ss * __expf(m - M); }
  }
  float lse  = M + __logf(S);
  float loss = (labels[t] >= 0) ? (lse - gold[t]) : 0.0f;

  __shared__ float red[256];
  red[threadIdx.x] = loss;
  __syncthreads();
  #pragma unroll
  for (int off = 128; off > 0; off >>= 1) {
    if (threadIdx.x < off) red[threadIdx.x] += red[threadIdx.x + off];
    __syncthreads();
  }
  if (threadIdx.x == 0) bsum[blockIdx.x] = red[0];
}

__global__ void finalize(const float* __restrict__ bsum,
                         const float* __restrict__ lw,
                         float* __restrict__ out) {
  float v = (threadIdx.x < 16) ? bsum[threadIdx.x] : 0.0f;
  #pragma unroll
  for (int s = 8; s > 0; s >>= 1) v += __shfl_xor(v, s, 32);
  if (threadIdx.x == 0) out[0] = v * lw[0];
}

// ---------------------------------------------------------------------------
extern "C" void kernel_launch(void* const* d_in, const int* in_sizes, int n_in,
                              void* d_out, int out_size, void* d_ws, size_t ws_size,
                              hipStream_t stream) {
  const float* H      = (const float*)d_in[0];   // [1,4096,2048] f32
  const float* W      = (const float*)d_in[1];   // [32000,2048]  f32
  const int*   labels = (const int*)  d_in[2];   // [1,4096]      i32
  const float* lw     = (const float*)d_in[3];   // [1]           f32
  float*       out    = (float*)d_out;

  char* ws = (char*)d_ws;
  constexpr size_t WB_OFF = 0;
  constexpr size_t HB_OFF = WB_OFF + (size_t)VOCAB * DIM * 2;
  constexpr size_t PM_OFF = HB_OFF + (size_t)T_TOK * DIM * 2;
  constexpr size_t PS_OFF = PM_OFF + (size_t)NSLAB * T_TOK * 4;
  constexpr size_t GD_OFF = PS_OFF + (size_t)NSLAB * T_TOK * 4;
  constexpr size_t BS_OFF = GD_OFF + (size_t)T_TOK * 4;

  bf16*  Wb   = (bf16*) (ws + WB_OFF);
  bf16*  Hb   = (bf16*) (ws + HB_OFF);
  float* pm   = (float*)(ws + PM_OFF);
  float* ps   = (float*)(ws + PS_OFF);
  float* gold = (float*)(ws + GD_OFF);
  float* bsum = (float*)(ws + BS_OFF);

  int nW4 = VOCAB * DIM / 4, nH4 = T_TOK * DIM / 4;
  cvt_f32_to_bf16<<<(nW4 + 255) / 256, 256, 0, stream>>>(W, Wb, nW4);
  cvt_f32_to_bf16<<<(nH4 + 255) / 256, 256, 0, stream>>>(H, Hb, nH4);

  gold_kernel<<<T_TOK / 8, 256, 0, stream>>>(H, W, labels, gold);

  dim3 grid(NSLAB, T_TOK / 128);
  gemm_lse_partial<<<grid, 256, 0, stream>>>(Hb, Wb, pm, ps);

  reduce_lse<<<T_TOK / 256, 256, 0, stream>>>(pm, ps, gold, labels, bsum);
  finalize<<<1, 32, 0, stream>>>(bsum, lw, out);
}
